// LNNASR_4552665333882
// MI455X (gfx1250) — matrix operand order repaired and checked by reading, not compiled, not run on hardware
//
#include <hip/hip_runtime.h>

// ---------- types ----------
typedef __attribute__((ext_vector_type(16))) __bf16       v16bf;
typedef __attribute__((ext_vector_type(8)))  float        v8f;
typedef __attribute__((ext_vector_type(4)))  unsigned int u32x4;

union BF16x16 { v16bf v; u32x4 q[2]; unsigned short s[16]; };

__device__ __forceinline__ unsigned short f2bf(float f) {
  unsigned int u = __float_as_uint(f);
  u += 0x7FFFu + ((u >> 16) & 1u);
  return (unsigned short)(u >> 16);
}
__device__ __forceinline__ float bf2f(unsigned short s) {
  return __uint_as_float(((unsigned int)s) << 16);
}
__device__ __forceinline__ float gelu_f(float x) {
  return 0.5f * x * (1.0f + erff(x * 0.7071067811865476f));
}
// branchless tanh on the scan critical path: v_exp_f32 + v_rcp_f32
__device__ __forceinline__ float tanh_fast(float x) {
  x = fminf(fmaxf(x, -9.0f), 9.0f);
  float e = __expf(2.0f * x);
  return (e - 1.0f) * __builtin_amdgcn_rcpf(e + 1.0f);
}
// async global->LDS 16B copy (GLOBAL_LOAD_ASYNC_TO_LDS_B128, tracked by ASYNCcnt)
__device__ __forceinline__ void async_copy_b128(void* lds_dst, const void* gsrc) {
  unsigned ldsOff = (unsigned)(size_t)lds_dst;   // low 32 bits of generic ptr = LDS offset
  asm volatile("global_load_async_to_lds_b128 %0, %1, off"
               :: "v"(ldsOff), "v"(gsrc) : "memory");
}
__device__ __forceinline__ void async_wait0() {
  asm volatile("s_wait_asynccnt 0x0" ::: "memory");
}

// ---------- weight repack: fp32 row-major [256, Nsrc] -> WMMA B-operand bf16 ----------
// element e = (((nt*8 + kc)*32 + lane)*16 + j)
//   k = kc*32 + (lane>>4)*16 + j ; n = nt*16 + (lane&15)
__global__ void ltc_pack_b(const float* __restrict__ W, unsigned short* __restrict__ dst,
                           int Npad, int Nsrc) {
  int e = blockIdx.x * 256 + threadIdx.x;
  if (e >= 256 * Npad) return;
  int nt = e >> 12;
  int r  = e & 4095;
  int kc = r >> 9;
  int r2 = r & 511;
  int L  = r2 >> 4;
  int j  = r2 & 15;
  int k  = kc * 32 + ((L >> 4) << 4) + j;
  int n  = nt * 16 + (L & 15);
  float val = (n < Nsrc) ? W[(size_t)k * Nsrc + n] : 0.0f;
  dst[e] = f2bf(val);
}

// ---------- input projection: gelu(LN(mel @ inW + inb)) -> bf16, [T,B,H] layout ----------
// grid (T, B): blockIdx.x = t, blockIdx.y = b
__global__ void __launch_bounds__(256)
ltc_input_proj(const float* __restrict__ mel, const float* __restrict__ inW,
               const float* __restrict__ inb, const float* __restrict__ g,
               const float* __restrict__ bb, unsigned short* __restrict__ x0, int T, int B) {
  __shared__ float melrow[80];
  __shared__ float red[256];
  __shared__ float smu, svar;
  const int t = blockIdx.x, b = blockIdx.y;
  const int h = threadIdx.x;
  size_t inrow  = (size_t)b * T + t;   // mel is [B,T,F]
  size_t outrow = (size_t)t * B + b;   // activations are [T,B,H]
  if (h < 80) melrow[h] = mel[inrow * 80 + h];
  __syncthreads();
  float s = inb[h];
#pragma unroll 4
  for (int f = 0; f < 80; ++f) s += melrow[f] * inW[f * 256 + h];
  red[h] = s; __syncthreads();
  for (int off = 128; off > 0; off >>= 1) { if (h < off) red[h] += red[h + off]; __syncthreads(); }
  if (h == 0) smu = red[0] * (1.0f / 256.0f);
  __syncthreads();
  float d = s - smu;
  red[h] = d * d; __syncthreads();
  for (int off = 128; off > 0; off >>= 1) { if (h < off) red[h] += red[h + off]; __syncthreads(); }
  if (h == 0) svar = red[0] * (1.0f / 256.0f);
  __syncthreads();
  float xn = d * rsqrtf(svar + 1e-5f) * g[h] + bb[h];
  x0[outrow * 256 + h] = f2bf(gelu_f(xn));
}

// ---------- LayerNorm over bf16 rows -> bf16 (row-layout agnostic) ----------
__global__ void __launch_bounds__(256)
ltc_layernorm(const unsigned short* __restrict__ x, const float* __restrict__ g,
              const float* __restrict__ bb, unsigned short* __restrict__ out) {
  __shared__ float red[256];
  __shared__ float smu, svar;
  size_t row = blockIdx.x;
  int h = threadIdx.x;
  float s = bf2f(x[row * 256 + h]);
  red[h] = s; __syncthreads();
  for (int off = 128; off > 0; off >>= 1) { if (h < off) red[h] += red[h + off]; __syncthreads(); }
  if (h == 0) smu = red[0] * (1.0f / 256.0f);
  __syncthreads();
  float d = s - smu;
  red[h] = d * d; __syncthreads();
  for (int off = 128; off > 0; off >>= 1) { if (h < off) red[h] += red[h + off]; __syncthreads(); }
  if (h == 0) svar = red[0] * (1.0f / 256.0f);
  __syncthreads();
  out[row * 256 + h] = f2bf(d * rsqrtf(svar + 1e-5f) * g[h] + bb[h]);
}

// ---------- bf16 WMMA GEMM: X[M,256] @ Wpack[256,Npad] + bias ----------
// block = 32*(Npad/16) threads, grid = M/64 ; each wave owns one 16-col stripe.
// A-tile staged to LDS with GLOBAL_LOAD_ASYNC_TO_LDS_B128 (ASYNCcnt), row-padded.
// outB != null : gelu -> bf16 (ld 256). else : f32 store, cols < outN, stride ldout;
// remap != 0 : row m (= t*64+b) is un-permuted to b*remapT + t on store (B=64).
__global__ void __launch_bounds__(512)
ltc_gemm(const unsigned short* __restrict__ X, const unsigned short* __restrict__ Wp,
         const float* __restrict__ bias, float* __restrict__ outF,
         unsigned short* __restrict__ outB, int outN, int ldout, int remap, int remapT) {
  __shared__ unsigned short Xs[64][264];  // padded rows: 528B stride -> bank-disjoint A reads
  const int tid = threadIdx.x, wave = tid >> 5, lane = tid & 31;
  const int nt = wave;
  const int n  = nt * 16 + (lane & 15);
  const size_t mbase = (size_t)blockIdx.x * 64;

  // async-stage the 64x256 bf16 A block directly into LDS (no VGPR round-trip)
  const unsigned short* Xblk = X + mbase * 256;
  for (int i = tid; i < 64 * 32; i += blockDim.x) {
    int r = i >> 5, c = i & 31;
    async_copy_b128(&Xs[r][c * 8], Xblk + r * 256 + c * 8);
  }

  BF16x16 breg[8];
#pragma unroll
  for (int kc = 0; kc < 8; ++kc) {
    const u32x4* p = (const u32x4*)(Wp + (((size_t)nt * 8 + kc) * 32 + lane) * 16);
    breg[kc].q[0] = p[0]; breg[kc].q[1] = p[1];
  }
  float bv = (bias && n < outN) ? bias[n] : 0.0f;
  async_wait0();
  __syncthreads();

  const int arow  = lane & 15;
  const int khalf = (lane >> 4) * 16;
#pragma unroll
  for (int mt = 0; mt < 4; ++mt) {
    BF16x16 a[8];
#pragma unroll
    for (int kc = 0; kc < 8; ++kc) {
      a[kc].q[0] = *(const u32x4*)&Xs[mt * 16 + arow][kc * 32 + khalf];
      a[kc].q[1] = *(const u32x4*)&Xs[mt * 16 + arow][kc * 32 + khalf + 8];
    }
    v8f acc0 = {0.f, 0.f, 0.f, 0.f, 0.f, 0.f, 0.f, 0.f};
    v8f acc1 = {0.f, 0.f, 0.f, 0.f, 0.f, 0.f, 0.f, 0.f};
#pragma unroll
    for (int kc = 0; kc < 8; kc += 2) {
      acc0 = __builtin_amdgcn_wmma_f32_16x16x32_bf16(false, a[kc].v,     false, breg[kc].v,
                                                     (short)0, acc0, false, false);
      acc1 = __builtin_amdgcn_wmma_f32_16x16x32_bf16(false, a[kc + 1].v, false, breg[kc + 1].v,
                                                     (short)0, acc1, false, false);
    }
#pragma unroll
    for (int v = 0; v < 8; ++v) {
      size_t m = mbase + mt * 16 + v + 8 * (lane >> 4);
      float val = acc0[v] + acc1[v] + bv;
      if (outB) {
        outB[m * 256 + n] = f2bf(gelu_f(val));
      } else if (n < outN) {
        size_t drow = remap ? ((size_t)(m & 63) * remapT + (m >> 6)) : m;
        outF[drow * (size_t)ldout + n] = val;
      }
    }
  }
}

// ---------- LTC scan: h_t = (h + f*A) * rcp(1 + 1/tau + f), f = tanh(px_t + h@Wh) ----------
// px, y in [T,B,H] layout (contiguous 16KB window per block per step).
// grid = B/16 blocks; 16 waves; Wh column stripe resident in VGPRs; h in LDS (bf16) + C regs.
__global__ void __launch_bounds__(512)
ltc_scan(const float* __restrict__ px, const unsigned short* __restrict__ Whp,
         const float* __restrict__ tau, const float* __restrict__ Avec,
         unsigned short* __restrict__ y, int T, int B) {
  __shared__ unsigned short hbuf[16][264];  // padded rows -> bank-disjoint A reads
  const int tid = threadIdx.x, wave = tid >> 5, lane = tid & 31;
  const int n  = wave * 16 + (lane & 15);
  const int b0 = blockIdx.x * 16;

  for (int i = tid; i < 16 * 264 / 2; i += blockDim.x) ((unsigned int*)hbuf)[i] = 0u;

  BF16x16 breg[8];
#pragma unroll
  for (int kc = 0; kc < 8; ++kc) {
    const u32x4* p = (const u32x4*)(Whp + (((size_t)wave * 8 + kc) * 32 + lane) * 16);
    breg[kc].q[0] = p[0]; breg[kc].q[1] = p[1];
  }
  const float a_n = Avec[n];
  const float dc  = 1.0f + 1.0f / tau[n];   // 1 + 1/tau
  v8f h = {0.f, 0.f, 0.f, 0.f, 0.f, 0.f, 0.f, 0.f};
  const int arow  = lane & 15;
  const int khalf = (lane >> 4) * 16;
  const int mhi   = 8 * (lane >> 4);
  __syncthreads();

  // software-pipelined px: preload t = 0
  float pc[8];
#pragma unroll
  for (int v = 0; v < 8; ++v)
    pc[v] = px[((size_t)(b0 + v + mhi)) * 256 + n];

  for (int t = 0; t < T; ++t) {
    // issue next-step px loads (double buffer) + L2 prefetch two steps ahead
    const int tn = (t + 1 < T) ? t + 1 : T - 1;
    const int t2 = (t + 2 < T) ? t + 2 : T - 1;
    float pn[8];
#pragma unroll
    for (int v = 0; v < 8; ++v)
      pn[v] = px[((size_t)tn * B + b0 + v + mhi) * 256 + n];
    __builtin_prefetch(px + ((size_t)t2 * B + b0) * 256 + tid * 8, 0, 1);

    // preload all 8 A-operands (16 ds_load_b128 back-to-back), then dual WMMA chains
    BF16x16 a[8];
#pragma unroll
    for (int kc = 0; kc < 8; ++kc) {
      a[kc].q[0] = *(const u32x4*)&hbuf[arow][kc * 32 + khalf];
      a[kc].q[1] = *(const u32x4*)&hbuf[arow][kc * 32 + khalf + 8];
    }
    v8f acc0 = {0.f, 0.f, 0.f, 0.f, 0.f, 0.f, 0.f, 0.f};
    v8f acc1 = {0.f, 0.f, 0.f, 0.f, 0.f, 0.f, 0.f, 0.f};
#pragma unroll
    for (int kc = 0; kc < 8; kc += 2) {
      acc0 = __builtin_amdgcn_wmma_f32_16x16x32_bf16(false, a[kc].v,     false, breg[kc].v,
                                                     (short)0, acc0, false, false);
      acc1 = __builtin_amdgcn_wmma_f32_16x16x32_bf16(false, a[kc + 1].v, false, breg[kc + 1].v,
                                                     (short)0, acc1, false, false);
    }
    unsigned short hb[8];
#pragma unroll
    for (int v = 0; v < 8; ++v) {
      float f  = tanh_fast(pc[v] + acc0[v] + acc1[v]);
      float hn = (h[v] + f * a_n) * __builtin_amdgcn_rcpf(dc + f);
      h[v]  = hn;
      hb[v] = f2bf(hn);
    }
    __syncthreads();  // all waves done reading h_{t-1}
#pragma unroll
    for (int v = 0; v < 8; ++v) {
      int m = v + mhi;
      hbuf[m][n] = hb[v];
      y[((size_t)t * B + b0 + m) * 256 + n] = hb[v];
    }
    __syncthreads();  // h_t visible to all waves
#pragma unroll
    for (int v = 0; v < 8; ++v) pc[v] = pn[v];
  }
}

__global__ void ltc_copy_lens(const unsigned long long* __restrict__ src,
                              unsigned long long* __restrict__ dst, int n) {
  int i = blockIdx.x * 64 + threadIdx.x;
  if (i < n) dst[i] = src[i];
}

// ---------- host launcher ----------
extern "C" void kernel_launch(void* const* d_in, const int* in_sizes, int n_in,
                              void* d_out, int out_size, void* d_ws, size_t ws_size,
                              hipStream_t stream) {
  (void)in_sizes; (void)n_in;
  const int B = 64, T = 2000, H = 256, Cc = 72;
  const size_t M = (size_t)B * T;

  const float* mel  = (const float*)d_in[0];
  const unsigned long long* lens = (const unsigned long long*)d_in[1];
  const float* inW  = (const float*)d_in[2];
  const float* inb  = (const float*)d_in[3];
  const float* ln1g = (const float*)d_in[4];
  const float* ln1b = (const float*)d_in[5];
  const float* Wx   = (const float*)d_in[6];
  const float* Wh   = (const float*)d_in[7];
  const float* lb   = (const float*)d_in[8];
  const float* tau  = (const float*)d_in[9];
  const float* Av   = (const float*)d_in[10];
  const float* outg = (const float*)d_in[11];
  const float* outb = (const float*)d_in[12];
  const float* W1   = (const float*)d_in[13];
  const float* b1   = (const float*)d_in[14];
  const float* W2   = (const float*)d_in[15];
  const float* b2   = (const float*)d_in[16];

  // workspace layout
  const size_t need = M * H * 2 * 2 + M * H * 4 + (5 * (size_t)H * H + (size_t)H * 80) * 2;
  if (ws_size < need) return;
  unsigned short* actA = (unsigned short*)d_ws;       // M*H bf16  ([T,B,H])
  unsigned short* actB = actA + M * H;                // M*H bf16  ([T,B,H])
  float*          px   = (float*)(actB + M * H);      // M*H f32   ([T,B,H])
  unsigned short* wx0p = (unsigned short*)(px + M * H);
  unsigned short* wx1p = wx0p + (size_t)H * H;
  unsigned short* wh0p = wx1p + (size_t)H * H;
  unsigned short* wh1p = wh0p + (size_t)H * H;
  unsigned short* w1p  = wh1p + (size_t)H * H;
  unsigned short* w2p  = w1p  + (size_t)H * H;        // H*80 (N padded 72->80)

  // pack weights into WMMA B-operand layout (bf16)
  ltc_pack_b<<<256, 256, 0, stream>>>(Wx,         wx0p, 256, 256);
  ltc_pack_b<<<256, 256, 0, stream>>>(Wx + H * H, wx1p, 256, 256);
  ltc_pack_b<<<256, 256, 0, stream>>>(Wh,         wh0p, 256, 256);
  ltc_pack_b<<<256, 256, 0, stream>>>(Wh + H * H, wh1p, 256, 256);
  ltc_pack_b<<<256, 256, 0, stream>>>(W1,         w1p,  256, 256);
  ltc_pack_b<<<80,  256, 0, stream>>>(W2,         w2p,   80,  72);

  // x0 = gelu(LN(mel @ inW + inb))  -> [T,B,H]
  ltc_input_proj<<<dim3(T, B), 256, 0, stream>>>(mel, inW, inb, ln1g, ln1b, actA, T, B);

  // layer 1
  ltc_gemm<<<(unsigned)(M / 64), 512, 0, stream>>>(actA, wx0p, lb,     px, nullptr, 256, 256, 0, T);
  ltc_scan<<<B / 16, 512, 0, stream>>>(px, wh0p, tau,     Av,     actB, T, B);
  // layer 2
  ltc_gemm<<<(unsigned)(M / 64), 512, 0, stream>>>(actB, wx1p, lb + H, px, nullptr, 256, 256, 0, T);
  ltc_scan<<<B / 16, 512, 0, stream>>>(px, wh1p, tau + H, Av + H, actA, T, B);

  // head: LN -> gelu(enc@W1+b1) -> @W2+b2 (store un-permuted to [B,T,C])
  ltc_layernorm<<<(unsigned)M, 256, 0, stream>>>(actA, outg, outb, actB);
  ltc_gemm<<<(unsigned)(M / 64), 512, 0, stream>>>(actB, w1p, b1, nullptr, actA, 256, 256, 0, T);
  ltc_gemm<<<(unsigned)(M / 64), 160, 0, stream>>>(actA, w2p, b2, (float*)d_out, nullptr, Cc, Cc, 1, T);

  // append mel_lengths after logits if the output buffer has room (int64 as 2 floats each)
  if ((size_t)out_size >= M * Cc + 2 * (size_t)B) {
    ltc_copy_lens<<<1, 64, 0, stream>>>(lens, (unsigned long long*)((float*)d_out + M * Cc), B);
  }
}
